// VQVAE_33638183863040
// MI455X (gfx1250) — matrix operand order repaired and checked by reading, not compiled, not run on hardware
//
#include <hip/hip_runtime.h>
#include <hip/hip_bf16.h>
#include <stdint.h>
#include <stddef.h>

typedef __attribute__((ext_vector_type(16))) _Float16 v16h;
typedef __attribute__((ext_vector_type(8)))  float    v8f;
typedef __attribute__((ext_vector_type(2)))  float    v2f;
typedef __attribute__((ext_vector_type(4)))  unsigned int u32x4;
typedef __attribute__((ext_vector_type(4)))  int      i32x4;
typedef __attribute__((ext_vector_type(8)))  int      i32x8;

// ---------------- problem dimensions ----------------
#define N_CONV1 (64*32*63*63)       // 8128512
#define N_CONV2 (64*64*32*32)       // 4194304
#define NTOK    65536               // 64*32*32
#define DIM     64
#define NCODE   512
#define N_ZP    (NTOK*DIM)          // 4194304
#define N_CT1   (64*64*64*64)       // 16777216
#define N_CT2   (64*32*128*128)     // 33554432
#define N_CT3   (64*3*125*125)      // 3000000

// ---------------- TDM helper: 2D row-major tile (global -> LDS) ----------------
// D# group0: count=1, lds_addr, global_addr(57b), type=2("image")
// D# group1: data_size=4B, tensor==tile (width x rows), stride0 = width (contiguous)
__device__ __forceinline__ void tdm_load_2d_f32(const float* gsrc, unsigned int lds_off,
                                                unsigned int width, unsigned int rows)
{
#if __has_builtin(__builtin_amdgcn_tensor_load_to_lds)
    unsigned long long ga = (unsigned long long)(uintptr_t)gsrc;
    u32x4 g0;
    g0[0] = 1u;                                   // count=1, user mode, no gather
    g0[1] = lds_off;                              // lds_addr (bytes)
    g0[2] = (unsigned int)ga;                     // global_addr[31:0]
    g0[3] = (unsigned int)(ga >> 32) | (2u << 30);// global_addr[56:32] | type=2
    i32x8 g1;
    g1[0] = (int)(2u << 16);                      // workgroup_mask=0, data_size=2 (4B)
    g1[1] = (int)((width & 0xFFFFu) << 16);       // tensor_dim0[15:0]
    g1[2] = (int)(((width >> 16) & 0xFFFFu) | ((rows & 0xFFFFu) << 16)); // dim0 hi | dim1 lo
    g1[3] = (int)((width & 0xFFFFu) << 16);       // tensor_dim1 hi = 0 | tile_dim0
    g1[4] = (int)(rows & 0xFFFFu);                // tile_dim1 | tile_dim2=0
    g1[5] = (int)width;                           // tensor_dim0_stride[31:0]
    g1[6] = 0;                                    // stride0 hi | stride1 lo
    g1[7] = 0;                                    // stride1 hi
    i32x4 z4 = {0, 0, 0, 0};
#if defined(__clang_major__) && (__clang_major__ >= 23)
    i32x8 z8 = {0, 0, 0, 0, 0, 0, 0, 0};
    __builtin_amdgcn_tensor_load_to_lds(g0, g1, z4, z4, z8, 0);
#else
    __builtin_amdgcn_tensor_load_to_lds(g0, g1, z4, z4, 0);
#endif
    __builtin_amdgcn_s_wait_tensorcnt(0);
#else
    (void)gsrc; (void)lds_off; (void)width; (void)rows;
#endif
}

// ---------------- misc ----------------
__global__ void init_accum_kernel(float* acc) {
    if (threadIdx.x < 8) acc[threadIdx.x] = 0.0f;
}

// ---------------- encoder ----------------
// conv1: x[64,3,125,125] -> y[64,32,63,63], k3 s2 p1, relu (3 in-ch: not matrix shaped)
__global__ __launch_bounds__(256) void conv1_kernel(
    const float* __restrict__ x, const float* __restrict__ w,
    const float* __restrict__ bias, float* __restrict__ y)
{
    int i = blockIdx.x * blockDim.x + threadIdx.x;
    if (i >= N_CONV1) return;
    int ow = i % 63; int t = i / 63;
    int oh = t % 63; t /= 63;
    int oc = t % 32; int b = t / 32;
    float acc = bias[oc];
    #pragma unroll
    for (int ic = 0; ic < 3; ++ic)
        for (int kh = 0; kh < 3; ++kh) {
            int ih = oh*2 - 1 + kh;
            if ((unsigned)ih >= 125u) continue;
            for (int kw = 0; kw < 3; ++kw) {
                int iw = ow*2 - 1 + kw;
                if ((unsigned)iw >= 125u) continue;
                acc += x[((size_t)(b*3+ic)*125 + ih)*125 + iw] *
                       w[((oc*3+ic)*3 + kh)*3 + kw];
            }
        }
    y[i] = fmaxf(acc, 0.0f);
}

// ------------- weight repack: make B-fragment loads contiguous over ic -------------
// forward conv weight w[oc][ic][kh][kw] -> p[(t*OCH+oc)*ICH+ic]
template<int ICH, int OCH>
__global__ __launch_bounds__(256) void repack_conv_kernel(
    const float* __restrict__ w, float* __restrict__ p)
{
    int i = blockIdx.x * blockDim.x + threadIdx.x;
    if (i >= OCH*ICH*9) return;
    int oc = i / (ICH*9); int r = i % (ICH*9);
    int ic = r / 9;       int t = r % 9;
    p[((size_t)t*OCH + oc)*ICH + ic] = w[i];
}

// convT weight w[ic][oc][kh][kw] -> p[(t*OCH+oc)*ICH+ic]
template<int ICH, int OCH>
__global__ __launch_bounds__(256) void repack_convT_kernel(
    const float* __restrict__ w, float* __restrict__ p)
{
    int i = blockIdx.x * blockDim.x + threadIdx.x;
    if (i >= ICH*OCH*9) return;
    int ic = i / (OCH*9); int r = i % (OCH*9);
    int oc = r / 9;       int t = r % 9;
    p[((size_t)t*OCH + oc)*ICH + ic] = w[i];
}

// ------------- conv2 implicit GEMM (f32 WMMA + TDM weight staging) -------------
// z1[64,32,63,63] -> y[64,64,32,32], k3 s2 p1, relu
// M = 16 consecutive ow, N = 16 oc, K = 9 taps x 32 ic.
// Packed weights (72KB) are DMA'd once per workgroup into LDS by the TDM;
// B fragments then come from LDS, keeping the VMEM pipe for activation gathers.
__global__ __launch_bounds__(256) void conv2_wmma_kernel(
    const float* __restrict__ in, const float* __restrict__ wp,
    const float* __restrict__ bias, float* __restrict__ out)
{
    constexpr int ICH = 32, OCH = 64, IHW = 63;
    __shared__ float smw[9*OCH*ICH];       // 18432 floats = 72KB

#if __has_builtin(__builtin_amdgcn_tensor_load_to_lds)
    if (threadIdx.x < 32) {                // one wave issues the TDM op and waits on it
        tdm_load_2d_f32(wp, (unsigned int)(uintptr_t)smw, OCH*ICH, 9);
    }
#else
    for (int i = threadIdx.x; i < 9*OCH*ICH; i += 256) smw[i] = wp[i];
#endif
    __syncthreads();

    int gtid = blockIdx.x * blockDim.x + threadIdx.x;
    int wave = gtid >> 5;                  // 64*32*2*4 = 16384 waves
    int lane = threadIdx.x & 31;
    int m = lane & 15, half = lane >> 4;
    int ocT = wave & 3;  int t1 = wave >> 2;
    int owT = t1 & 1;    t1 >>= 1;
    int oh  = t1 & 31;   int b = t1 >> 5;
    int ow = owT*16 + m;                   // A row (output pixel)
    int oc = ocT*16 + m;                   // B col (output channel)
    const float* inb = in + (size_t)b*ICH*IHW*IHW;
    v8f acc = {};
    #pragma unroll
    for (int t = 0; t < 9; ++t) {
        int kh = t/3, kw = t%3;
        int ih = oh*2 - 1 + kh;
        int iw = ow*2 - 1 + kw;
        bool valid = ((unsigned)ih < (unsigned)IHW) && ((unsigned)iw < (unsigned)IHW);
        int ihc = min(max(ih, 0), IHW-1);
        int iwc = min(max(iw, 0), IHW-1);
        const float* abase = inb + (size_t)ihc*IHW + iwc;
        const float* bbase = smw + ((size_t)t*OCH + oc)*ICH;
        #pragma unroll 2
        for (int j = 0; j < ICH/4; ++j) {
            int ic = j*4 + half*2;
            float v0 = abase[(size_t)ic*IHW*IHW];       // always in-bounds
            float v1 = abase[(size_t)(ic+1)*IHW*IHW];
            v2f a, bb;
            a[0] = valid ? v0 : 0.0f;
            a[1] = valid ? v1 : 0.0f;
            bb[0] = bbase[ic];
            bb[1] = bbase[ic+1];
            acc = __builtin_amdgcn_wmma_f32_16x16x4_f32(false, a, false, bb,
                                                        (short)0, acc, false, false);
        }
    }
    float bval = bias[oc];
    // C vgpr r: row M = half*8+r (ow in tile), col N = m (oc in tile)
    float* orow = out + ((size_t)(b*OCH + ocT*16 + m))*1024 + (size_t)oh*32 + owT*16 + half*8;
    #pragma unroll
    for (int r = 0; r < 8; ++r) orow[r] = fmaxf(acc[r] + bval, 0.0f);
}

// ------------- conv3 (1x1) GEMM (f32 WMMA) -------------
// z2[64,64,32,32] -> zp[NTOK,64] + f16 copy; M = 16 tokens, N = 16 d, K = 64 ic
__global__ __launch_bounds__(256) void conv3_wmma_kernel(
    const float* __restrict__ z2, const float* __restrict__ w,
    const float* __restrict__ bias, float* __restrict__ zp,
    _Float16* __restrict__ zp_h)
{
    int gtid = blockIdx.x * blockDim.x + threadIdx.x;
    int wave = gtid >> 5;                  // 4096*4 = 16384 waves
    int lane = threadIdx.x & 31;
    int m = lane & 15, half = lane >> 4;
    int colT = wave & 3;
    int n0 = (wave >> 2) * 16;
    int d0 = colT * 16;
    int b  = n0 >> 10;
    int hw = (n0 & 1023) + m;              // A row: token n0+m
    const float* zb = z2 + (size_t)b*64*1024;
    int d = d0 + m;                        // B col
    v8f acc = {};
    #pragma unroll 2
    for (int k0 = 0; k0 < 64; k0 += 4) {
        int c = k0 + half*2;
        v2f a, bb;
        a[0] = zb[(size_t)c*1024 + hw];
        a[1] = zb[(size_t)(c+1)*1024 + hw];
        bb[0] = w[(size_t)d*64 + c];       // w[d][c], contiguous pair
        bb[1] = w[(size_t)d*64 + c + 1];
        acc = __builtin_amdgcn_wmma_f32_16x16x4_f32(false, a, false, bb,
                                                    (short)0, acc, false, false);
    }
    float bval = bias[d];
    #pragma unroll
    for (int r = 0; r < 8; ++r) {
        int n = n0 + half*8 + r;
        float v = acc[r] + bval;
        zp[(size_t)n*64 + d]   = v;
        zp_h[(size_t)n*64 + d] = (_Float16)v;
    }
}

// ---------------- VQ ----------------
__global__ __launch_bounds__(256) void cbprep_kernel(
    const float* __restrict__ cb, _Float16* __restrict__ cb_h,
    float* __restrict__ cnorm)
{
    int k = blockIdx.x * blockDim.x + threadIdx.x;
    if (k >= NCODE) return;
    float s = 0.0f;
    for (int d = 0; d < DIM; ++d) {
        _Float16 h = (_Float16)cb[k*DIM + d];
        cb_h[k*DIM + d] = h;
        float f = (float)h;
        s += f * f;
    }
    cnorm[k] = s;
}

// score(n,k) = |c_k|^2 - 2 z_n.c_k  (|z_n|^2 constant per row); f16 WMMA, K=64 in 2 steps
__global__ __launch_bounds__(256) void vq_argmin_kernel(
    const _Float16* __restrict__ zp_h, const _Float16* __restrict__ cb_h,
    const float* __restrict__ cnorm, int* __restrict__ idx)
{
    const int gtid = blockIdx.x * blockDim.x + threadIdx.x;
    const int wave = gtid >> 5;            // 4096 waves
    const int lane = threadIdx.x & 31;
    const int m    = lane & 15;
    const int half = lane >> 4;
    const int row0 = wave * 16;

    const _Float16* arow = zp_h + (size_t)(row0 + m) * DIM;
    v16h a0, a1;
    #pragma unroll
    for (int j = 0; j < 8; ++j) {
        a0[j]     = arow[half*8 + j];
        a0[8 + j] = arow[16 + half*8 + j];
        a1[j]     = arow[32 + half*8 + j];
        a1[8 + j] = arow[48 + half*8 + j];
    }

    float minv[8];
    int   mini[8];
    #pragma unroll
    for (int r = 0; r < 8; ++r) { minv[r] = 3.4e38f; mini[r] = 0; }

    for (int t = 0; t < NCODE/16; ++t) {
        const int code = t*16 + m;
        const _Float16* brow = cb_h + (size_t)code * DIM;
        v16h b0, b1;
        #pragma unroll
        for (int e = 0; e < 8; ++e) {
            b0[e]     = brow[half*8 + e];
            b0[8 + e] = brow[16 + half*8 + e];
            b1[e]     = brow[32 + half*8 + e];
            b1[8 + e] = brow[48 + half*8 + e];
        }
        v8f acc = {};
        acc = __builtin_amdgcn_wmma_f32_16x16x32_f16(false, a0, false, b0,
                                                     (short)0, acc, false, false);
        acc = __builtin_amdgcn_wmma_f32_16x16x32_f16(false, a1, false, b1,
                                                     (short)0, acc, false, false);
        const float cn = cnorm[code];
        #pragma unroll
        for (int r = 0; r < 8; ++r) {
            float score = cn - 2.0f * acc[r];
            if (score < minv[r]) { minv[r] = score; mini[r] = code; }
        }
    }
    #pragma unroll
    for (int off = 1; off < 16; off <<= 1) {
        #pragma unroll
        for (int r = 0; r < 8; ++r) {
            float ov = __shfl_xor(minv[r], off, 32);
            int   oi = __shfl_xor(mini[r], off, 32);
            if (ov < minv[r] || (ov == minv[r] && oi < mini[r])) {
                minv[r] = ov; mini[r] = oi;
            }
        }
    }
    if (m == 0) {
        #pragma unroll
        for (int r = 0; r < 8; ++r) idx[row0 + half*8 + r] = mini[r];
    }
}

// gather q, exact fp32 SSE, emit e = q in NCHW
__global__ __launch_bounds__(256) void vq_gather_kernel(
    const float* __restrict__ zp, const float* __restrict__ cb,
    const int* __restrict__ idx, float* __restrict__ e,
    float* __restrict__ vq_sse)
{
    int i = blockIdx.x * blockDim.x + threadIdx.x;
    float local = 0.0f;
    if (i < N_ZP) {
        int d = i & 63; int n = i >> 6;
        int b = n >> 10; int hw = n & 1023;
        float q = cb[(size_t)idx[n]*DIM + d];
        float diff = q - zp[i];
        local = diff * diff;
        e[((size_t)b*64 + d)*1024 + hw] = q;
    }
    __shared__ float red[256];
    red[threadIdx.x] = local;
    __syncthreads();
    for (int s = 128; s > 0; s >>= 1) {
        if ((int)threadIdx.x < s) red[threadIdx.x] += red[threadIdx.x + s];
        __syncthreads();
    }
    if (threadIdx.x == 0) atomicAdd(vq_sse, red[0]);
}

// ------------- parity-decomposed convT implicit GEMM (f32 WMMA) -------------
// stride2 pad1 opad1: in[64][ICH][IHW][IHW] -> out[64][OCH][2*IHW][2*IHW], relu
//   py==0 -> kh=1 only; py==1 -> kh in {0,2}   (same for px/kw)
// K = {1,2}x{1,2}xICH instead of 9xICH. M = 16 same-parity columns, N = 16 oc.
template<int ICH, int OCH, int IHW>
__global__ __launch_bounds__(256) void convT_wmma_kernel(
    const float* __restrict__ in, const float* __restrict__ wp,
    const float* __restrict__ bias, float* __restrict__ out)
{
    constexpr int OHW = IHW*2;
    constexpr int WTILES = OHW/32;         // tiles of 16 same-parity columns
    constexpr int CTILES = OCH/16;
    int gtid = blockIdx.x * blockDim.x + threadIdx.x;
    int wave = gtid >> 5;
    int lane = threadIdx.x & 31;
    int m = lane & 15, half = lane >> 4;

    int cls = wave & 3;      int t1 = wave >> 2;
    int ocT = t1 % CTILES;   t1 /= CTILES;
    int owT = t1 % WTILES;   t1 /= WTILES;
    int ohh = t1 % (OHW/2);  int b = t1 / (OHW/2);
    int py = cls >> 1, px = cls & 1;
    int oh = ohh*2 + py;
    int ow = owT*32 + 2*m + px;            // A row (same-parity output pixel)
    int oc = ocT*16 + m;                   // B col

    const float* inb = in + (size_t)b*ICH*IHW*IHW;
    const int nkh = py ? 2 : 1;
    const int nkw = px ? 2 : 1;
    v8f acc = {};
    for (int ki = 0; ki < nkh; ++ki) {
        int kh = py ? ki*2 : 1;
        int ih = (oh + 1 - kh) >> 1;       // exact by parity
        bool rv = (unsigned)ih < (unsigned)IHW;   // fails only kh=0, oh=OHW-1
        int ihc = min(ih, IHW-1);
        for (int wi = 0; wi < nkw; ++wi) {
            int kw = px ? wi*2 : 1;
            int iw = (ow + 1 - kw) >> 1;   // exact by parity, >= 0
            bool valid = rv && (iw < IHW); // fails only kw=0, ow=OHW-1
            int iwc = min(iw, IHW-1);
            int t = kh*3 + kw;
            const float* abase = inb + (size_t)ihc*IHW + iwc;
            const float* bbase = wp + ((size_t)t*OCH + oc)*ICH;
            #pragma unroll 2
            for (int j = 0; j < ICH/4; ++j) {
                int ic = j*4 + half*2;
                float v0 = abase[(size_t)ic*IHW*IHW];
                float v1 = abase[(size_t)(ic+1)*IHW*IHW];
                v2f a, bb;
                a[0] = valid ? v0 : 0.0f;
                a[1] = valid ? v1 : 0.0f;
                bb[0] = bbase[ic];
                bb[1] = bbase[ic+1];
                acc = __builtin_amdgcn_wmma_f32_16x16x4_f32(false, a, false, bb,
                                                            (short)0, acc, false, false);
            }
        }
    }
    float bval = bias[oc];
    // C vgpr r: row M = half*8+r -> ow = owT*32 + 2*(half*8+r) + px, col N = m -> oc
    float* orow = out + ((size_t)(b*OCH + ocT*16 + m))*OHW*OHW
                      + (size_t)oh*OHW + owT*32 + px + 2*(half*8);
    #pragma unroll
    for (int r = 0; r < 8; ++r) orow[2*r] = fmaxf(acc[r] + bval, 0.0f);
}

// convT3: h2[64,32,128,128] -> x_recon[64,3,125,125], w[in32,out3,2,2], s1 p2 + recon SSE
__global__ __launch_bounds__(256) void convT3_kernel(
    const float* __restrict__ h2, const float* __restrict__ w,
    const float* __restrict__ bias, const float* __restrict__ x,
    float* __restrict__ xrec, float* __restrict__ rec_sse)
{
    int i = blockIdx.x * blockDim.x + threadIdx.x;
    float local = 0.0f;
    if (i < N_CT3) {
        int ow = i % 125; int t = i / 125;
        int oh = t % 125; t /= 125;
        int oc = t % 3;  int b = t / 3;
        float acc = bias[oc];
        #pragma unroll
        for (int kh = 0; kh < 2; ++kh) {
            int ih = oh + 2 - kh;
            #pragma unroll
            for (int kw = 0; kw < 2; ++kw) {
                int iw = ow + 2 - kw;
                const float* hin = h2 + (size_t)b*32*16384 + (size_t)ih*128 + iw;
                const float* wk  = w + oc*4 + kh*2 + kw;
                float s = 0.0f;
                #pragma unroll 8
                for (int ic = 0; ic < 32; ++ic)
                    s += hin[(size_t)ic*16384] * wk[ic*12];
                acc += s;
            }
        }
        xrec[i] = acc;
        float diff = acc - x[i];
        local = diff * diff;
    }
    __shared__ float red[256];
    red[threadIdx.x] = local;
    __syncthreads();
    for (int s = 128; s > 0; s >>= 1) {
        if ((int)threadIdx.x < s) red[threadIdx.x] += red[threadIdx.x + s];
        __syncthreads();
    }
    if (threadIdx.x == 0) atomicAdd(rec_sse, red[0]);
}

__global__ void finalize_kernel(const float* __restrict__ accum, float* __restrict__ out3) {
    if (threadIdx.x == 0) {
        float mse = accum[0] / (float)N_ZP;
        float eq  = mse + 0.25f * mse;
        float rec = accum[1];
        out3[0] = eq + rec;
        out3[1] = eq;
        out3[2] = rec;
    }
}

// ---------------- host launcher ----------------
static inline size_t align256(size_t v) { return (v + 255) & ~(size_t)255; }

extern "C" void kernel_launch(void* const* d_in, const int* in_sizes, int n_in,
                              void* d_out, int out_size, void* d_ws, size_t ws_size,
                              hipStream_t stream) {
    const float* x      = (const float*)d_in[0];
    const float* enc_w1 = (const float*)d_in[1];
    const float* enc_b1 = (const float*)d_in[2];
    const float* enc_w2 = (const float*)d_in[3];
    const float* enc_b2 = (const float*)d_in[4];
    const float* enc_w3 = (const float*)d_in[5];
    const float* enc_b3 = (const float*)d_in[6];
    const float* cb     = (const float*)d_in[7];
    const float* dec_w1 = (const float*)d_in[8];
    const float* dec_b1 = (const float*)d_in[9];
    const float* dec_w2 = (const float*)d_in[10];
    const float* dec_b2 = (const float*)d_in[11];
    const float* dec_w3 = (const float*)d_in[12];
    const float* dec_b3 = (const float*)d_in[13];

    char* ws = (char*)d_ws;
    size_t off = 0;
    float*     z1    = (float*)(ws + off); off = align256(off + sizeof(float)*N_CONV1);
    float*     z2    = (float*)(ws + off); off = align256(off + sizeof(float)*N_CONV2);
    float*     zp    = (float*)(ws + off); off = align256(off + sizeof(float)*N_ZP);
    _Float16*  zp_h  = (_Float16*)(ws + off); off = align256(off + sizeof(_Float16)*N_ZP);
    _Float16*  cb_h  = (_Float16*)(ws + off); off = align256(off + sizeof(_Float16)*NCODE*DIM);
    float*     cnorm = (float*)(ws + off); off = align256(off + sizeof(float)*NCODE);
    int*       idx   = (int*)(ws + off); off = align256(off + sizeof(int)*NTOK);
    float*     e     = (float*)(ws + off); off = align256(off + sizeof(float)*N_ZP);
    float*     h1    = (float*)(ws + off); off = align256(off + sizeof(float)*N_CT1);
    float*     h2    = (float*)(ws + off); off = align256(off + sizeof(float)*N_CT2);
    float*     wp2   = (float*)(ws + off); off = align256(off + sizeof(float)*9*64*32);
    float*     wpT1  = (float*)(ws + off); off = align256(off + sizeof(float)*9*64*64);
    float*     wpT2  = (float*)(ws + off); off = align256(off + sizeof(float)*9*32*64);
    float*     accum = (float*)(ws + off); off = align256(off + sizeof(float)*8);

    float* out  = (float*)d_out;   // [0]=loss_sum [1]=e_q_loss [2]=recon_loss
    float* xrec = out + 3;         // [64,3,125,125]

    const int T = 256;
    init_accum_kernel<<<1, 32, 0, stream>>>(accum);

    conv1_kernel<<<(N_CONV1 + T - 1)/T, T, 0, stream>>>(x, enc_w1, enc_b1, z1);

    repack_conv_kernel<32,64><<<(64*32*9 + T - 1)/T, T, 0, stream>>>(enc_w2, wp2);
    conv2_wmma_kernel<<<(16384*32)/T, T, 0, stream>>>(z1, wp2, enc_b2, z2);

    conv3_wmma_kernel<<<(16384*32)/T, T, 0, stream>>>(z2, enc_w3, enc_b3, zp, zp_h);

    cbprep_kernel<<<(NCODE + T - 1)/T, T, 0, stream>>>(cb, cb_h, cnorm);
    vq_argmin_kernel<<<(4096*32)/T, T, 0, stream>>>(zp_h, cb_h, cnorm, idx);
    vq_gather_kernel<<<(N_ZP + T - 1)/T, T, 0, stream>>>(zp, cb, idx, e, accum + 0);

    repack_convT_kernel<64,64><<<(64*64*9 + T - 1)/T, T, 0, stream>>>(dec_w1, wpT1);
    convT_wmma_kernel<64,64,32><<<(65536*32)/T, T, 0, stream>>>(e, wpT1, dec_b1, h1);   // 64*32*2*4*4 waves

    repack_convT_kernel<64,32><<<(64*32*9 + T - 1)/T, T, 0, stream>>>(dec_w2, wpT2);
    convT_wmma_kernel<64,32,64><<<(131072*32)/T, T, 0, stream>>>(h1, wpT2, dec_b2, h2); // 64*64*4*2*4 waves

    convT3_kernel<<<(N_CT3 + T - 1)/T, T, 0, stream>>>(h2, dec_w3, dec_b3, x, xrec, accum + 1);
    finalize_kernel<<<1, 32, 0, stream>>>(accum, out);
    (void)in_sizes; (void)n_in; (void)out_size; (void)ws_size;
}
